// SparseHypergraphLayer_68667937129227
// MI455X (gfx1250) — matrix operand order repaired and checked by reading, not compiled, not run on hardware
//
#include <hip/hip_runtime.h>
#include <math.h>

typedef _Float16 v16h __attribute__((ext_vector_type(16)));
typedef _Float16 v8h  __attribute__((ext_vector_type(8)));
typedef float    v8f  __attribute__((ext_vector_type(8)));

#define N_NODES   50000
#define N_INC     300000
#define N_EDGES   20000
#define DIM       256
#define NEG_SLOPE 0.2f
#define BN_EPS    1e-5f
#define LDS_STRIDE 264   // 16B-aligned rows, 4-bank rotation per row -> conflict-free b128 LDS reads

// ---------------- order-preserving float<->uint encoding for atomicMax ----------------
__device__ __forceinline__ unsigned fenc(float f) {
  unsigned u = __float_as_uint(f);
  return (u & 0x80000000u) ? ~u : (u | 0x80000000u);
}
__device__ __forceinline__ float fdec(unsigned u) {
  u = (u & 0x80000000u) ? (u & 0x7fffffffu) : ~u;
  return __uint_as_float(u);
}

// ---------------- W [K,D] f32 -> Wt [D,K] f16 (B-matrix staging) ----------------
__global__ __launch_bounds__(256) void k_wt(const float* __restrict__ W, _Float16* __restrict__ Wt) {
  int idx = blockIdx.x * 256 + threadIdx.x;        // 65536 elements
  int k = idx >> 8, d = idx & 255;
  Wt[d * DIM + k] = (_Float16)W[k * DIM + d];
}

// ---------------- GEMM: xw = x @ W via v_wmma_f32_16x16x32_f16 ----------------
// block = 128 threads (4 waves); block computes rows [16*b, 16*b+16) x all 256 cols.
// wave w owns cols [64w, 64w+64) = 4 WMMA tiles, reusing one A fragment.
__global__ __launch_bounds__(128) void k_gemm(const float* __restrict__ x,
                                              const _Float16* __restrict__ Wt,
                                              float* __restrict__ xw) {
  __shared__ _Float16 lds[16 * LDS_STRIDE];
  const int row0 = blockIdx.x * 16;
  const int tid  = threadIdx.x;

  // stage A tile (16 rows x 256 k), converting f32 -> f16 once per element
  for (int i = tid; i < 16 * DIM; i += 128) {
    int r = i >> 8, k = i & 255;
    lds[r * LDS_STRIDE + k] = (_Float16)x[(size_t)(row0 + r) * DIM + k];
  }
  __syncthreads();

  const int wave = tid >> 5, lane = tid & 31;
  const int m = lane & 15, h8 = lane >> 4;
  const int col0 = wave * 64;

  v8f acc0 = {}, acc1 = {}, acc2 = {}, acc3 = {};

  for (int kk = 0; kk < DIM; kk += 32) {
    // A fragment (16x32 f16): lane holds row m; half j -> k = kk + h8*16 + j
    // (contiguous order, matching B)
    v8h alo = *(const v8h*)&lds[m * LDS_STRIDE + kk + h8 * 16];
    v8h ahi = *(const v8h*)&lds[m * LDS_STRIDE + kk + h8 * 16 + 8];
    v16h a = __builtin_shufflevector(alo, ahi, 0,1,2,3,4,5,6,7,8,9,10,11,12,13,14,15);

    // B fragment (32x16 f16): lane holds col n = colbase + m; half j -> k = kk + h8*16 + j
    const _Float16* wb = Wt + (size_t)(col0 + m) * DIM + kk + h8 * 16;
    v16h b0 = *(const v16h*)(wb);
    v16h b1 = *(const v16h*)(wb + 16 * DIM);
    v16h b2 = *(const v16h*)(wb + 32 * DIM);
    v16h b3 = *(const v16h*)(wb + 48 * DIM);

    acc0 = __builtin_amdgcn_wmma_f32_16x16x32_f16(false, a, false, b0, (short)0, acc0, false, false);
    acc1 = __builtin_amdgcn_wmma_f32_16x16x32_f16(false, a, false, b1, (short)0, acc1, false, false);
    acc2 = __builtin_amdgcn_wmma_f32_16x16x32_f16(false, a, false, b2, (short)0, acc2, false, false);
    acc3 = __builtin_amdgcn_wmma_f32_16x16x32_f16(false, a, false, b3, (short)0, acc3, false, false);
  }

  // C/D layout: VGPR r -> M = h8*8 + r, N = m
  for (int r = 0; r < 8; ++r) {
    float* o = xw + (size_t)(row0 + h8 * 8 + r) * DIM + col0 + m;
    o[0] = acc0[r]; o[16] = acc1[r]; o[32] = acc2[r]; o[48] = acc3[r];
  }
}

// ---------------- per-node attention dots: s1 = xw . att[:256], s2 = xw . att[256:] ----------------
__global__ __launch_bounds__(256) void k_dots(const float* __restrict__ xw, const float* __restrict__ att,
                                              float* __restrict__ s1, float* __restrict__ s2) {
  int wave = threadIdx.x >> 5, lane = threadIdx.x & 31;
  int node = blockIdx.x * 8 + wave;
  if (node >= N_NODES) return;
  const float* row = xw + (size_t)node * DIM;
  float a1 = 0.f, a2 = 0.f;
#pragma unroll
  for (int j = 0; j < 8; ++j) {
    int d = lane + j * 32;
    float v = row[d];
    a1 += v * att[d];
    a2 += v * att[DIM + d];
  }
#pragma unroll
  for (int off = 16; off > 0; off >>= 1) {
    a1 += __shfl_xor(a1, off, 32);
    a2 += __shfl_xor(a2, off, 32);
  }
  if (lane == 0) { s1[node] = a1; s2[node] = a2; }
}

// ---------------- alpha (leaky-relu) + degrees + segment max ----------------
__global__ __launch_bounds__(256) void k_alpha(const int* __restrict__ nidx, const int* __restrict__ eidx,
                                               const float* __restrict__ s1, const float* __restrict__ s2,
                                               float* __restrict__ alpha, float* __restrict__ Dn,
                                               float* __restrict__ Be, unsigned* __restrict__ amax) {
  int e = blockIdx.x * 256 + threadIdx.x;
  if (e >= N_INC) return;
  int ni = nidx[e], ei = eidx[e];
  float a = s1[ni] + s2[ei];
  a = (a > 0.f) ? a : NEG_SLOPE * a;
  alpha[e] = a;
  atomicAdd(&Dn[ni], 1.0f);
  atomicAdd(&Be[ei], 1.0f);
  atomicMax(&amax[ni], fenc(a));
}

// ---------------- exp(alpha - amax) + segment sum ----------------
__global__ __launch_bounds__(256) void k_exp(const int* __restrict__ nidx, float* __restrict__ alpha,
                                             const unsigned* __restrict__ amax, float* __restrict__ esum) {
  int e = blockIdx.x * 256 + threadIdx.x;
  if (e >= N_INC) return;
  int ni = nidx[e];
  float ex = expf(alpha[e] - fdec(amax[ni]));
  alpha[e] = ex;
  atomicAdd(&esum[ni], ex);
}

// ---------------- final softmax weight + stage coefficients ----------------
__global__ __launch_bounds__(256) void k_coef(const int* __restrict__ nidx, const int* __restrict__ eidx,
                                              const float* __restrict__ alpha, const float* __restrict__ esum,
                                              const float* __restrict__ Dn, const float* __restrict__ Be,
                                              float* __restrict__ coef1, float* __restrict__ coef2) {
  int e = blockIdx.x * 256 + threadIdx.x;
  if (e >= N_INC) return;
  int ni = nidx[e], ei = eidx[e];
  float a = alpha[e] / (esum[ni] + 1e-16f);
  coef1[e] = a / Be[ei];   // Binv at target edge (Be>=1 here by construction)
  coef2[e] = a / Dn[ni];   // Dinv at target node (Dn>=1 here by construction)
}

// ---------------- stage 1: node -> hyperedge scatter (L2-resident atomics) ----------------
__global__ __launch_bounds__(256) void k_stage1(const int* __restrict__ nidx, const int* __restrict__ eidx,
                                                const float* __restrict__ coef1, const float* __restrict__ xw,
                                                float* __restrict__ edge_feat) {
  int e = blockIdx.x * 8 + (threadIdx.x >> 5);
  if (e >= N_INC) return;
  int lane = threadIdx.x & 31;
  int ni = nidx[e], ei = eidx[e];
  float c = coef1[e];
  const float* src = xw + (size_t)ni * DIM;
  float* dst = edge_feat + (size_t)ei * DIM;
#pragma unroll
  for (int j = 0; j < 8; ++j) {
    int d = lane + j * 32;
    atomicAdd(&dst[d], c * src[d]);
  }
}

// ---------------- stage 2: hyperedge -> node scatter into d_out ----------------
__global__ __launch_bounds__(256) void k_stage2(const int* __restrict__ nidx, const int* __restrict__ eidx,
                                                const float* __restrict__ coef2, const float* __restrict__ edge_feat,
                                                float* __restrict__ out) {
  int e = blockIdx.x * 8 + (threadIdx.x >> 5);
  if (e >= N_INC) return;
  int lane = threadIdx.x & 31;
  int ni = nidx[e], ei = eidx[e];
  float c = coef2[e];
  const float* src = edge_feat + (size_t)ei * DIM;
  float* dst = out + (size_t)ni * DIM;
#pragma unroll
  for (int j = 0; j < 8; ++j) {
    int d = lane + j * 32;
    atomicAdd(&dst[d], c * src[d]);
  }
}

// ---------------- BN stats: coalesced per-dim partial sums (thread d <-> dim d) ----------------
#define ROWS_PER_BLOCK 200   // 250 blocks * 200 rows = 50000
__global__ __launch_bounds__(256) void k_stats(const float* __restrict__ out, const float* __restrict__ bias,
                                               float* __restrict__ sum, float* __restrict__ sumsq) {
  int d = threadIdx.x;
  int r0 = blockIdx.x * ROWS_PER_BLOCK;
  float b = bias[d], s = 0.f, ss = 0.f;
  for (int r = 0; r < ROWS_PER_BLOCK; ++r) {
    float v = out[(size_t)(r0 + r) * DIM + d] + b;
    s += v; ss += v * v;
  }
  atomicAdd(&sum[d], s);
  atomicAdd(&sumsq[d], ss);
}

// ---------------- BN normalize + ELU, in place on d_out ----------------
__global__ __launch_bounds__(256) void k_norm(float* __restrict__ out, const float* __restrict__ bias,
                                              const float* __restrict__ gamma, const float* __restrict__ beta,
                                              const float* __restrict__ sum, const float* __restrict__ sumsq) {
  size_t idx = (size_t)blockIdx.x * 256 + threadIdx.x;   // grid = N_NODES blocks -> d = threadIdx.x
  int d = threadIdx.x;
  float mu  = sum[d]   * (1.0f / N_NODES);
  float var = sumsq[d] * (1.0f / N_NODES) - mu * mu;
  float v = out[idx] + bias[d];
  v = gamma[d] * (v - mu) * rsqrtf(var + BN_EPS) + beta[d];
  out[idx] = (v > 0.f) ? v : expm1f(v);
}

extern "C" void kernel_launch(void* const* d_in, const int* in_sizes, int n_in,
                              void* d_out, int out_size, void* d_ws, size_t ws_size,
                              hipStream_t stream) {
  const float* x        = (const float*)d_in[0];
  const int*   node_idx = (const int*)d_in[1];
  const int*   edge_idx = (const int*)d_in[2];
  // d_in[3] = num_hyperedges scalar (M = 20000, compile-time constant here)
  const float* W        = (const float*)d_in[4];
  const float* att      = (const float*)d_in[5];
  const float* bias     = (const float*)d_in[6];
  const float* gamma    = (const float*)d_in[7];
  const float* beta     = (const float*)d_in[8];
  float* out = (float*)d_out;

  char* ws = (char*)d_ws;
  size_t off = 0;
  auto carve = [&](size_t bytes) -> void* {
    void* p = ws + off;
    off += (bytes + 255) & ~(size_t)255;
    return p;
  };

  float*     xw        = (float*)carve((size_t)N_NODES * DIM * 4);
  _Float16*  Wt        = (_Float16*)carve((size_t)DIM * DIM * 2);
  float*     s1        = (float*)carve((size_t)N_NODES * 4);
  float*     s2        = (float*)carve((size_t)N_NODES * 4);
  float*     alpha     = (float*)carve((size_t)N_INC * 4);
  float*     coef1     = (float*)carve((size_t)N_INC * 4);
  float*     coef2     = (float*)carve((size_t)N_INC * 4);
  unsigned*  amax      = (unsigned*)carve((size_t)N_NODES * 4);
  float*     esum      = (float*)carve((size_t)N_NODES * 4);
  float*     Dn        = (float*)carve((size_t)N_NODES * 4);
  float*     Be        = (float*)carve((size_t)N_EDGES * 4);
  float*     edge_feat = (float*)carve((size_t)N_EDGES * DIM * 4);
  float*     dsum      = (float*)carve((size_t)DIM * 4);
  float*     dsumsq    = (float*)carve((size_t)DIM * 4);
  (void)ws_size; (void)in_sizes; (void)n_in; (void)out_size;

  // zero all accumulators (graph-capture-safe async memsets)
  hipMemsetAsync(amax,      0, (size_t)N_NODES * 4, stream);   // enc==0 < enc(any finite) -> valid -inf seed
  hipMemsetAsync(esum,      0, (size_t)N_NODES * 4, stream);
  hipMemsetAsync(Dn,        0, (size_t)N_NODES * 4, stream);
  hipMemsetAsync(Be,        0, (size_t)N_EDGES * 4, stream);
  hipMemsetAsync(edge_feat, 0, (size_t)N_EDGES * DIM * 4, stream);
  hipMemsetAsync(dsum,      0, (size_t)DIM * 4, stream);
  hipMemsetAsync(dsumsq,    0, (size_t)DIM * 4, stream);
  hipMemsetAsync(out,       0, (size_t)N_NODES * DIM * 4, stream);

  // 1) stage W as f16 transposed
  k_wt<<<(DIM * DIM) / 256, 256, 0, stream>>>(W, Wt);

  // 2) xw = x @ W  (WMMA f16 -> f32)
  k_gemm<<<N_NODES / 16, 128, 0, stream>>>(x, Wt, xw);

  // 3) per-node attention dot products
  k_dots<<<N_NODES / 8, 256, 0, stream>>>(xw, att, s1, s2);

  // 4) attention logits, degrees, segment max
  const int gE = (N_INC + 255) / 256;
  k_alpha<<<gE, 256, 0, stream>>>(node_idx, edge_idx, s1, s2, alpha, Dn, Be, amax);

  // 5) exp + segment sum
  k_exp<<<gE, 256, 0, stream>>>(node_idx, alpha, amax, esum);

  // 6) softmax normalize + degree coefficients
  k_coef<<<gE, 256, 0, stream>>>(node_idx, edge_idx, alpha, esum, Dn, Be, coef1, coef2);

  // 7) node -> hyperedge
  k_stage1<<<N_INC / 8, 256, 0, stream>>>(node_idx, edge_idx, coef1, xw, edge_feat);

  // 8) hyperedge -> node (accumulate into d_out)
  k_stage2<<<N_INC / 8, 256, 0, stream>>>(node_idx, edge_idx, coef2, edge_feat, out);

  // 9) BN stats, 10) BN + ELU
  k_stats<<<N_NODES / ROWS_PER_BLOCK, 256, 0, stream>>>(out, bias, dsum, dsumsq);
  k_norm<<<N_NODES, 256, 0, stream>>>(out, bias, gamma, beta, dsum, dsumsq);
}
